// GraphSAGE_63393717289265
// MI455X (gfx1250) — matrix operand order repaired and checked by reading, not compiled, not run on hardware
//
#include <hip/hip_runtime.h>

typedef __attribute__((ext_vector_type(16))) _Float16 v16h;
typedef __attribute__((ext_vector_type(8)))  float    v8f;

#define IN_FEATS 512
#define HIDDEN   128
#define NEG_SLOPE 0.01f
#define APAD 34   // LDS row stride in halves (breaks bank conflicts, keeps 4B align)

// ---------------- utility kernels ----------------

__global__ void k_zero(float* __restrict__ p, int n) {
  int i = blockIdx.x * blockDim.x + threadIdx.x;
  if (i < n) p[i] = 0.0f;
}

__global__ void k_deg(const int* __restrict__ dst, float* __restrict__ deg, int nE) {
  int i = blockIdx.x * blockDim.x + threadIdx.x;
  if (i < nE) atomicAdd(&deg[dst[i]], 1.0f);
}

__global__ void k_inv(float* __restrict__ deg, int n) {
  int i = blockIdx.x * blockDim.x + threadIdx.x;
  if (i < n) deg[i] = 1.0f / fmaxf(deg[i], 1.0f);
}

// one wave32 per edge: lane handles 4 of 128 feats (float4 gather + 4 atomics)
__global__ void k_scatter(const float* __restrict__ h, const int* __restrict__ src,
                          const int* __restrict__ dst, float* __restrict__ hn, int nE) {
  int wave = (blockIdx.x * blockDim.x + threadIdx.x) >> 5;
  int lane = threadIdx.x & 31;
  if (wave >= nE) return;
  int s = src[wave];
  int d = dst[wave];
  const float4 v = *(const float4*)(h + (size_t)s * HIDDEN + lane * 4);
  float* o = hn + (size_t)d * HIDDEN + lane * 4;
  atomicAdd(o + 0, v.x);
  atomicAdd(o + 1, v.y);
  atomicAdd(o + 2, v.z);
  atomicAdd(o + 3, v.w);
}

// ---------------- WMMA fragment helpers ----------------
// A (16x32 f16): lane L -> row M = L%16; elements 0..7 hold K = (L<16?0..7:8..15),
// elements 8..15 hold K = (L<16?16..23:24..31).
// B (32x16 f16): lane L -> col N = L%16; same K layout as A.

__device__ __forceinline__ v16h load_a_frag(const _Float16* sA, int wv, int lane) {
  v16h a;
  int m  = (wv << 4) + (lane & 15);
  int hi = (lane >> 4) << 3;  // 0 or 8
  const _Float16* row = sA + m * APAD;
#pragma unroll
  for (int e = 0; e < 8; ++e) a[e] = row[hi + e];
#pragma unroll
  for (int e = 0; e < 8; ++e) a[e + 8] = row[hi + 16 + e];
  return a;
}

__device__ __forceinline__ v16h load_b_frag(const _Float16* sB, int t, int lane) {
  v16h b;
  int n  = (t << 4) + (lane & 15);
  int hi = (lane >> 4) << 3;  // 0 or 8
  const _Float16* row = sB + n * APAD;
#pragma unroll
  for (int e = 0; e < 8; ++e) b[e] = row[hi + e];
#pragma unroll
  for (int e = 0; e < 8; ++e) b[e + 8] = row[hi + 16 + e];
  return b;
}

// ---------------- input projection: h = X @ W^T + b ----------------
// X: [nRows, 512], W: [128, 512] (torch convention), Y: [nRows, 128]
__global__ __launch_bounds__(256) void k_gemm_in(const float* __restrict__ X,
                                                 const float* __restrict__ W,
                                                 const float* __restrict__ bias,
                                                 float* __restrict__ Y, int nRows) {
  __shared__ _Float16 sA[128 * APAD];
  __shared__ _Float16 sB[128 * APAD];
  const int tid  = threadIdx.x;
  const int lane = tid & 31;
  const int wv   = tid >> 5;
  const int row0 = blockIdx.x * 128;

  v8f acc[8];
#pragma unroll
  for (int t = 0; t < 8; ++t) acc[t] = (v8f){0, 0, 0, 0, 0, 0, 0, 0};

  for (int k0 = 0; k0 < IN_FEATS; k0 += 32) {
    for (int i = tid; i < 128 * 32; i += 256) {
      int r = i >> 5, k = i & 31;
      int gr = row0 + r;
      float v = (gr < nRows) ? X[(size_t)gr * IN_FEATS + k0 + k] : 0.0f;
      sA[r * APAD + k] = (_Float16)v;
    }
    for (int i = tid; i < 128 * 32; i += 256) {
      int n = i >> 5, k = i & 31;
      sB[n * APAD + k] = (_Float16)W[(size_t)n * IN_FEATS + k0 + k];
    }
    __syncthreads();

    v16h a = load_a_frag(sA, wv, lane);
#pragma unroll
    for (int t = 0; t < 8; ++t) {
      v16h b = load_b_frag(sB, t, lane);
      acc[t] = __builtin_amdgcn_wmma_f32_16x16x32_f16(false, a, false, b,
                                                      (short)0, acc[t], false, false);
    }
    __syncthreads();
  }

  // C/D layout: lane L -> col N = L%16; VGPR v -> row M = v + (L<16?0:8)
  const int lane15 = lane & 15;
  const int rIn = (wv << 4) + ((lane >> 4) << 3);
#pragma unroll
  for (int t = 0; t < 8; ++t) {
    int col = (t << 4) + lane15;
    float bc = bias[col];
#pragma unroll
    for (int v = 0; v < 8; ++v) {
      int gr = row0 + rIn + v;
      if (gr < nRows) Y[(size_t)gr * HIDDEN + col] = acc[t][v] + bc;
    }
  }
}

// ---------------- SAGE layer: h = leaky(h @ Ws^T + bs + (hn*inv_deg) @ Wn^T) ----------------
__global__ __launch_bounds__(256) void k_layer(float* __restrict__ h,
                                               const float* __restrict__ hn,
                                               const float* __restrict__ invd,
                                               const float* __restrict__ Wself,
                                               const float* __restrict__ bself,
                                               const float* __restrict__ Wneigh,
                                               int nRows) {
  __shared__ _Float16 sA[128 * APAD];
  __shared__ _Float16 sB[128 * APAD];
  const int tid  = threadIdx.x;
  const int lane = tid & 31;
  const int wv   = tid >> 5;
  const int row0 = blockIdx.x * 128;

  v8f acc[8];
#pragma unroll
  for (int t = 0; t < 8; ++t) acc[t] = (v8f){0, 0, 0, 0, 0, 0, 0, 0};

  for (int pass = 0; pass < 2; ++pass) {
    const float* A = pass ? hn : h;
    const float* W = pass ? Wneigh : Wself;
    for (int k0 = 0; k0 < HIDDEN; k0 += 32) {
      for (int i = tid; i < 128 * 32; i += 256) {
        int r = i >> 5, k = i & 31;
        int gr = row0 + r;
        float v = 0.0f;
        if (gr < nRows) {
          v = A[(size_t)gr * HIDDEN + k0 + k];
          if (pass) v *= invd[gr];
        }
        sA[r * APAD + k] = (_Float16)v;
      }
      for (int i = tid; i < 128 * 32; i += 256) {
        int n = i >> 5, k = i & 31;
        sB[n * APAD + k] = (_Float16)W[(size_t)n * HIDDEN + k0 + k];
      }
      __syncthreads();

      v16h a = load_a_frag(sA, wv, lane);
#pragma unroll
      for (int t = 0; t < 8; ++t) {
        v16h b = load_b_frag(sB, t, lane);
        acc[t] = __builtin_amdgcn_wmma_f32_16x16x32_f16(false, a, false, b,
                                                        (short)0, acc[t], false, false);
      }
      __syncthreads();
    }
  }

  const int lane15 = lane & 15;
  const int rIn = (wv << 4) + ((lane >> 4) << 3);
#pragma unroll
  for (int t = 0; t < 8; ++t) {
    int col = (t << 4) + lane15;
    float bc = bself[col];
#pragma unroll
    for (int v = 0; v < 8; ++v) {
      int gr = row0 + rIn + v;
      if (gr < nRows) {
        float val = acc[t][v] + bc;
        val = (val > 0.0f) ? val : val * NEG_SLOPE;
        h[(size_t)gr * HIDDEN + col] = val;  // safe: block writes only rows it read
      }
    }
  }
}

// ---------------- output projection: out = h @ Wout^T + bout ----------------
__global__ void k_out(const float* __restrict__ h, const float* __restrict__ Wout,
                      const float* __restrict__ bout, float* __restrict__ out,
                      int nRows, int nClasses) {
  int i = blockIdx.x * blockDim.x + threadIdx.x;
  if (i >= nRows) return;
  const float4* hr = (const float4*)(h + (size_t)i * HIDDEN);
  for (int c = 0; c < nClasses; ++c) {
    const float4* wr = (const float4*)(Wout + (size_t)c * HIDDEN);
    float s = bout[c];
#pragma unroll 8
    for (int f = 0; f < HIDDEN / 4; ++f) {
      float4 a = hr[f], b = wr[f];
      s += a.x * b.x + a.y * b.y + a.z * b.z + a.w * b.w;
    }
    out[(size_t)i * nClasses + c] = s;
  }
}

// ---------------- launcher ----------------
extern "C" void kernel_launch(void* const* d_in, const int* in_sizes, int n_in,
                              void* d_out, int out_size, void* d_ws, size_t ws_size,
                              hipStream_t stream) {
  const float* features = (const float*)d_in[0];
  const int*   src      = (const int*)d_in[1];
  const int*   dst      = (const int*)d_in[2];
  const float* W_in     = (const float*)d_in[3];
  const float* b_in     = (const float*)d_in[4];
  const float* W_self   = (const float*)d_in[5];
  const float* b_self   = (const float*)d_in[6];
  const float* W_neigh  = (const float*)d_in[7];
  const float* W_out    = (const float*)d_in[8];
  const float* b_out    = (const float*)d_in[9];
  float* out = (float*)d_out;

  const int nNodes   = in_sizes[0] / IN_FEATS;
  const int nEdges   = in_sizes[1];
  const int nClasses = in_sizes[8] / HIDDEN;
  const int nLayers  = in_sizes[5] / (HIDDEN * HIDDEN);

  float* ws  = (float*)d_ws;
  float* deg = ws;                                   // nNodes (becomes inv_deg)
  float* h   = ws + nNodes;                          // nNodes * 128
  float* hn  = h + (size_t)nNodes * HIDDEN;          // nNodes * 128

  const int nHF = nNodes * HIDDEN;
  const int nBlkGemm = (nNodes + 127) / 128;

  k_zero<<<(nNodes + 255) / 256, 256, 0, stream>>>(deg, nNodes);
  k_deg<<<(nEdges + 255) / 256, 256, 0, stream>>>(dst, deg, nEdges);
  k_inv<<<(nNodes + 255) / 256, 256, 0, stream>>>(deg, nNodes);

  k_gemm_in<<<nBlkGemm, 256, 0, stream>>>(features, W_in, b_in, h, nNodes);

  for (int l = 0; l < nLayers; ++l) {
    k_zero<<<(nHF + 255) / 256, 256, 0, stream>>>(hn, nHF);
    k_scatter<<<(nEdges + 7) / 8, 256, 0, stream>>>(h, src, dst, hn, nEdges);
    k_layer<<<nBlkGemm, 256, 0, stream>>>(h, hn, deg,
                                          W_self + (size_t)l * HIDDEN * HIDDEN,
                                          b_self + (size_t)l * HIDDEN,
                                          W_neigh + (size_t)l * HIDDEN * HIDDEN,
                                          nNodes);
  }

  k_out<<<(nNodes + 255) / 256, 256, 0, stream>>>(h, W_out, b_out, out, nNodes, nClasses);
}
